// RSSM_44753559224373
// MI455X (gfx1250) — matrix operand order repaired and checked by reading, not compiled, not run on hardware
//
#include <hip/hip_runtime.h>

// ---------------------------------------------------------------------------
// RSSM scan for MI455X (gfx1250).
// GEMMs: bf16 operands, fp32 accumulate via v_wmma_f32_16x16x32_bf16.
// Operand tiles are staged global->LDS with GLOBAL_LOAD_ASYNC_TO_LDS_B128
// (ASYNCcnt / s_wait_asynccnt), triple-buffered, so each block reads every
// A/B element from L2 exactly once; waves read WMMA fragments from LDS.
// Weights (~56MB bf16) stay L2-resident across the 128 sequential timesteps.
// Workspace requirement: ~79 MB.
// ---------------------------------------------------------------------------

typedef __attribute__((ext_vector_type(16))) __bf16 v16bf;
typedef __attribute__((ext_vector_type(8)))  __bf16 v8bf;
typedef __attribute__((ext_vector_type(8)))  float  v8f;

#define B_   256
#define T_   128
#define H_   2048
#define L_   256
#define E_   1024
#define A_   32
#define NP_  256   // padded "200" dim (multiple of 128 -> no N guards anywhere)

// ---------------- helpers ----------------

__device__ __forceinline__ __bf16 f2bf(float f) {
  unsigned u = __builtin_bit_cast(unsigned, f);
  unsigned r = u + 0x7FFFu + ((u >> 16) & 1u);   // round-to-nearest-even
  unsigned short hs = (unsigned short)(r >> 16);
  return __builtin_bit_cast(__bf16, hs);
}

__device__ __forceinline__ v8f zero8f() {
  v8f z;
#pragma unroll
  for (int i = 0; i < 8; ++i) z[i] = 0.0f;
  return z;
}

union BF16x16 { v16bf v; v8bf h[2]; };

__device__ __forceinline__ v8f wmma_bf16(v16bf a, v16bf b, v8f c) {
  return __builtin_amdgcn_wmma_f32_16x16x32_bf16(
      /*neg_a=*/false, a, /*neg_b=*/false, b,
      /*c_mod=*/(short)0, c, /*reuse_a=*/false, /*reuse_b=*/false);
}

// Async 16-byte copy global -> LDS (tracked by ASYNCcnt).
__device__ __forceinline__ void async_cp16(unsigned lds_off, const __bf16* g) {
  asm volatile("global_load_async_to_lds_b128 %0, %1, off"
               :: "v"(lds_off), "v"(g) : "memory");
}

// ---------------- kernels ----------------

// Convert fp32 weight [N,K] -> zero-padded bf16 [Npad,Kpad] (row-major).
__global__ void convert_pad(const float* __restrict__ src, __bf16* __restrict__ dst,
                            int N, int K, int Npad, int Kpad) {
  long long i = (long long)blockIdx.x * blockDim.x + threadIdx.x;
  long long total = (long long)Npad * Kpad;
  if (i >= total) return;
  int n = (int)(i / Kpad), k = (int)(i % Kpad);
  float v = (n < N && k < K) ? src[(long long)n * K + k] : 0.0f;
  dst[i] = f2bf(v);
}

__global__ void copy_f32(const float* __restrict__ src, float* __restrict__ dst, int n) {
  int i = blockIdx.x * blockDim.x + threadIdx.x;
  if (i < n) dst[i] = src[i];
}

__global__ void cvt_f32_bf16(const float* __restrict__ src, __bf16* __restrict__ dst, int n) {
  int i = blockIdx.x * blockDim.x + threadIdx.x;
  if (i < n) dst[i] = f2bf(src[i]);
}

// dst[rows, w1+w2] = [ s1(bf16, stride w1) | cvt(s2 fp32, stride s2stride) ]
__global__ void cat_bf_f32(const __bf16* __restrict__ s1, int w1,
                           const float* __restrict__ s2, int s2stride, int w2,
                           __bf16* __restrict__ dst, int rows) {
  int W = w1 + w2;
  int i = blockIdx.x * blockDim.x + threadIdx.x;
  if (i >= rows * W) return;
  int r = i / W, c = i % W;
  dst[i] = (c < w1) ? s1[(size_t)r * w1 + c]
                    : f2bf(s2[(size_t)r * s2stride + (c - w1)]);
}

// dst[rows, w1+w2] = [ s1(bf16) | s2(bf16) ]
__global__ void cat_bf_bf(const __bf16* __restrict__ s1, int w1,
                          const __bf16* __restrict__ s2, int w2,
                          __bf16* __restrict__ dst, int rows) {
  int W = w1 + w2;
  int i = blockIdx.x * blockDim.x + threadIdx.x;
  if (i >= rows * W) return;
  int r = i / W, c = i % W;
  dst[i] = (c < w1) ? s1[(size_t)r * w1 + c] : s2[(size_t)r * w2 + (c - w1)];
}

// C[M x N] = act( A[M x K](bf16) * W[N x K]^T(bf16) + bias )
// Requirements: M % 128 == 0, N % 128 == 0, K % 32 == 0.
// Block: 256 threads = 8 waves in 4(M) x 2(N); block tile 128 x 128;
// wave tile 32 x 64 (2 A-frags x 4 B-frags -> 8 WMMA per 32-K step).
// Operands staged via async global->LDS copies, 3 buffers, 2 stages ahead.
#define LDW 40                        // padded LDS row stride (elems)
#define LSTRIDE (128 * LDW)           // one buffer (elems)
__global__ void __launch_bounds__(256)
gemm_bf16(const __bf16* __restrict__ A, const __bf16* __restrict__ W,
          int K, const float* __restrict__ bias, int Nbias,
          float* __restrict__ OutF, __bf16* __restrict__ OutB,
          int sOut, int act) {
  __shared__ __align__(16) __bf16 lsA[3][LSTRIDE];
  __shared__ __align__(16) __bf16 lsB[3][LSTRIDE];

  const int tid   = threadIdx.x;
  const int lane  = tid & 31;
  const int wid   = tid >> 5;
  const int wm    = wid >> 1;        // 0..3
  const int wn    = wid & 1;         // 0..1
  const int r     = lane & 15;
  const int khalf = lane >> 4;

  // Cooperative copy mapping: thread -> (tile row, 16-elem chunk); 32B/thread.
  const int rowCp = tid >> 1;                 // 0..127
  const int ckCp  = (tid & 1) * 16;           // 0 or 16 (elems)
  const __bf16* gA = A + (size_t)(blockIdx.x * 128 + rowCp) * K + ckCp;
  const __bf16* gB = W + (size_t)(blockIdx.y * 128 + rowCp) * K + ckCp;
  const unsigned laBase = (unsigned)(uintptr_t)&lsA[0][rowCp * LDW + ckCp];
  const unsigned lbBase = (unsigned)(uintptr_t)&lsB[0][rowCp * LDW + ckCp];

  auto issue = [&](int k0, int buf) {
    const unsigned la = laBase + (unsigned)buf * (LSTRIDE * 2);
    const unsigned lb = lbBase + (unsigned)buf * (LSTRIDE * 2);
    async_cp16(la,      gA + k0);
    async_cp16(la + 16, gA + k0 + 8);
    async_cp16(lb,      gB + k0);
    async_cp16(lb + 16, gB + k0 + 8);
  };

  v8f acc[2][4];
#pragma unroll
  for (int i = 0; i < 2; ++i)
#pragma unroll
    for (int j = 0; j < 4; ++j) acc[i][j] = zero8f();

  // Prologue: prefetch stages 0 and 1.
  issue(0, 0);
  if (K > 32) issue(32, 1);

  int s = 0;
  for (int k0 = 0; k0 < K; k0 += 32) {
    // Prefetch stage k+2 into the buffer freed at the end of iteration k-1.
    if (k0 + 64 < K) {
      int nb = s - 1; if (nb < 0) nb += 3;
      issue(k0 + 64, nb);
    }
    const int remain = (K - k0 - 32) / 32;    // stages issued beyond current
    if (remain >= 2)      asm volatile("s_wait_asynccnt 0x8" ::: "memory");
    else if (remain == 1) asm volatile("s_wait_asynccnt 0x4" ::: "memory");
    else                  asm volatile("s_wait_asynccnt 0x0" ::: "memory");
    __syncthreads();

    // Fragment reads from LDS.
    // A-frag: 16 contiguous K per lane at khalf*16 (two 16B chunks).
    // B-frag: 16 contiguous K per lane at khalf*16 (two 16B chunks).
    BF16x16 a0, a1, b0, b1, b2, b3;
    const __bf16* pa = &lsA[s][(wm * 32 + r) * LDW + khalf * 16];
    a0.h[0] = *(const v8bf*)pa;
    a0.h[1] = *(const v8bf*)(pa + 8);
    a1.h[0] = *(const v8bf*)(pa + 16 * LDW);
    a1.h[1] = *(const v8bf*)(pa + 16 * LDW + 8);
    const __bf16* pb = &lsB[s][(wn * 64 + r) * LDW + khalf * 16];
    b0.h[0] = *(const v8bf*)pb;
    b0.h[1] = *(const v8bf*)(pb + 8);
    b1.h[0] = *(const v8bf*)(pb + 16 * LDW);
    b1.h[1] = *(const v8bf*)(pb + 16 * LDW + 8);
    b2.h[0] = *(const v8bf*)(pb + 32 * LDW);
    b2.h[1] = *(const v8bf*)(pb + 32 * LDW + 8);
    b3.h[0] = *(const v8bf*)(pb + 48 * LDW);
    b3.h[1] = *(const v8bf*)(pb + 48 * LDW + 8);

    acc[0][0] = wmma_bf16(a0.v, b0.v, acc[0][0]);
    acc[0][1] = wmma_bf16(a0.v, b1.v, acc[0][1]);
    acc[0][2] = wmma_bf16(a0.v, b2.v, acc[0][2]);
    acc[0][3] = wmma_bf16(a0.v, b3.v, acc[0][3]);
    acc[1][0] = wmma_bf16(a1.v, b0.v, acc[1][0]);
    acc[1][1] = wmma_bf16(a1.v, b1.v, acc[1][1]);
    acc[1][2] = wmma_bf16(a1.v, b2.v, acc[1][2]);
    acc[1][3] = wmma_bf16(a1.v, b3.v, acc[1][3]);

    __syncthreads();
    s = s + 1; if (s == 3) s = 0;
  }

  const int mWave = blockIdx.x * 128 + wm * 32;
  const int nWave = blockIdx.y * 128 + wn * 64;

  // Epilogue: D layout -> lane gives N (lane&15) and M-half (lane>>4)*8.
#pragma unroll
  for (int i = 0; i < 2; ++i) {
#pragma unroll
    for (int j = 0; j < 4; ++j) {
      const int col = nWave + 16 * j + (lane & 15);
      const int rowBase = mWave + 16 * i + (lane >> 4) * 8;
      const float bv = (bias != nullptr && col < Nbias) ? bias[col] : 0.0f;
#pragma unroll
      for (int v = 0; v < 8; ++v) {
        float x = acc[i][j][v] + bv;
        if (act) x = fmaxf(x, 0.0f);
        const size_t o = (size_t)(rowBase + v) * sOut + col;
        if (OutF) OutF[o] = x;
        if (OutB) OutB[o] = f2bf(x);
      }
    }
  }
}

// GRU gates: gi/gh [256, 3H] with biases already added; h_f32 updated in
// place; also writes h_bf (for downstream GEMMs) and the output h[B,T,H].
__global__ void gru_gate(const float* __restrict__ gi, const float* __restrict__ gh,
                         float* __restrict__ h, __bf16* __restrict__ h_bf,
                         float* __restrict__ out_h, int t) {
  int idx = blockIdx.x * blockDim.x + threadIdx.x;
  if (idx >= B_ * H_) return;
  int m = idx >> 11;          // /2048
  int n = idx & (H_ - 1);
  const float* gim = gi + (size_t)m * (3 * H_);
  const float* ghm = gh + (size_t)m * (3 * H_);
  float rr = 1.0f / (1.0f + __expf(-(gim[n] + ghm[n])));
  float zz = 1.0f / (1.0f + __expf(-(gim[n + H_] + ghm[n + H_])));
  float ng = tanhf(gim[n + 2 * H_] + rr * ghm[n + 2 * H_]);
  float hp = h[idx];
  float hn = (1.0f - zz) * ng + zz * hp;
  h[idx]    = hn;
  h_bf[idx] = f2bf(hn);
  out_h[((size_t)m * T_ + t) * H_ + n] = hn;
}

// Heads epilogue: lin [256, 512] = [mu | std_pre]; writes mu, std, sample
// (fp32 outputs) and optionally a bf16 copy of the sample for downstream GEMMs.
__global__ void head_elem(const float* __restrict__ lin,
                          const float* __restrict__ bmu, const float* __restrict__ bstd,
                          const float* __restrict__ noise, int t,
                          float* __restrict__ out_mu, float* __restrict__ out_std,
                          float* __restrict__ out_s, __bf16* __restrict__ state_bf) {
  int idx = blockIdx.x * blockDim.x + threadIdx.x;
  if (idx >= B_ * L_) return;
  int m = idx >> 8;
  int n = idx & (L_ - 1);
  float mu = lin[(size_t)m * 512 + n] + bmu[n];
  float sx = lin[(size_t)m * 512 + 256 + n] + bstd[n];
  float sp = (sx > 20.0f) ? sx : log1pf(__expf(sx));
  float sd = sp + 0.1f;
  float nz = noise[((size_t)m * T_ + t) * L_ + n];
  float s  = mu + sd * nz;
  size_t o = ((size_t)m * T_ + t) * L_ + n;
  out_mu[o]  = mu;
  out_std[o] = sd;
  out_s[o]   = s;
  if (state_bf) state_bf[idx] = f2bf(s);
}

// Reward head: N=1, tiny — scalar dot per batch row (fp32 rf2).
__global__ void reward_dot(const float* __restrict__ rf2, const float* __restrict__ Wr3,
                           const float* __restrict__ br3, float* __restrict__ out_rew, int t) {
  int b = blockIdx.x * blockDim.x + threadIdx.x;
  if (b >= B_) return;
  const float* row = rf2 + (size_t)b * NP_;
  float s = br3[0];
  for (int k = 0; k < 200; ++k) s += row[k] * Wr3[k];
  out_rew[(size_t)b * T_ + t] = s;
}

// ---------------- host ----------------

extern "C" void kernel_launch(void* const* d_in, const int* in_sizes, int n_in,
                              void* d_out, int out_size, void* d_ws, size_t ws_size,
                              hipStream_t stream) {
  (void)in_sizes; (void)n_in; (void)out_size; (void)ws_size;

  // Inputs (setup_inputs order)
  const float* prevS = (const float*)d_in[0];
  const float* prevH = (const float*)d_in[1];
  const float* enc   = (const float*)d_in[2];
  const float* acts  = (const float*)d_in[3];
  const float* nzp   = (const float*)d_in[4];
  const float* nzq   = (const float*)d_in[5];
  const float* W_embed = (const float*)d_in[6];  const float* b_embed = (const float*)d_in[7];
  const float* W_ih  = (const float*)d_in[8];    const float* b_ih  = (const float*)d_in[9];
  const float* W_hh  = (const float*)d_in[10];   const float* b_hh  = (const float*)d_in[11];
  const float* Wp1 = (const float*)d_in[12];     const float* bp1 = (const float*)d_in[13];
  const float* Wp2 = (const float*)d_in[14];     const float* bp2 = (const float*)d_in[15];
  const float* Wpm = (const float*)d_in[16];     const float* bpm = (const float*)d_in[17];
  const float* Wps = (const float*)d_in[18];     const float* bps = (const float*)d_in[19];
  const float* Wq1 = (const float*)d_in[20];     const float* bq1 = (const float*)d_in[21];
  const float* Wq2 = (const float*)d_in[22];     const float* bq2 = (const float*)d_in[23];
  const float* Wqm = (const float*)d_in[24];     const float* bqm = (const float*)d_in[25];
  const float* Wqs = (const float*)d_in[26];     const float* bqs = (const float*)d_in[27];
  const float* Wr1 = (const float*)d_in[28];     const float* br1 = (const float*)d_in[29];
  const float* Wr2 = (const float*)d_in[30];     const float* br2 = (const float*)d_in[31];
  const float* Wr3 = (const float*)d_in[32];     const float* br3 = (const float*)d_in[33];

  // bf16 weight region in workspace (element offsets)
  __bf16* wsb = (__bf16*)d_ws;
  const size_t oWemb = 0;                              // 2048 x 288
  const size_t oWih  = oWemb + (size_t)2048 * 288;     // 6144 x 2048
  const size_t oWhh  = oWih  + (size_t)6144 * 2048;
  const size_t oWp1  = oWhh  + (size_t)6144 * 2048;    // 256 x 2048
  const size_t oWp2  = oWp1  + (size_t)NP_ * 2048;     // 256 x 256
  const size_t oWph  = oWp2  + (size_t)NP_ * NP_;      // 512 x 256 (Wpm|Wps)
  const size_t oWq1  = oWph  + (size_t)512 * NP_;      // 256 x 3072
  const size_t oWq2  = oWq1  + (size_t)NP_ * 3072;     // 256 x 256
  const size_t oWqh  = oWq2  + (size_t)NP_ * NP_;      // 512 x 256 (Wqm|Wqs)
  const size_t oWr1  = oWqh  + (size_t)512 * NP_;      // 256 x 2304
  const size_t oWr2  = oWr1  + (size_t)NP_ * 2304;     // 256 x 256
  const size_t wEnd  = oWr2  + (size_t)NP_ * NP_;

  // bf16 activation region
  __bf16* post_bf = wsb + wEnd;                        // 256 x 256
  __bf16* h_bf    = post_bf + (size_t)B_ * L_;         // 256 x 2048
  __bf16* emb_bf  = h_bf    + (size_t)B_ * H_;         // 256 x 2048
  __bf16* xcat    = emb_bf  + (size_t)B_ * H_;         // 256 x 288
  __bf16* qcat    = xcat    + (size_t)B_ * (L_ + A_);  // 256 x 3072
  __bf16* rcat    = qcat    + (size_t)B_ * (H_ + E_);  // 256 x 2304
  __bf16* pf1_bf  = rcat    + (size_t)B_ * (L_ + H_);  // 256 x 256
  __bf16* pf_bf   = pf1_bf  + (size_t)B_ * NP_;
  __bf16* qf1_bf  = pf_bf   + (size_t)B_ * NP_;
  __bf16* qf_bf   = qf1_bf  + (size_t)B_ * NP_;
  __bf16* rf1_bf  = qf_bf   + (size_t)B_ * NP_;
  __bf16* bfEnd   = rf1_bf  + (size_t)B_ * NP_;

  // fp32 region
  float* wsf = (float*)bfEnd;
  float* h_f  = wsf;                                   // 256 x 2048
  float* gi   = h_f  + (size_t)B_ * H_;                // 256 x 6144
  float* gh   = gi   + (size_t)B_ * 3 * H_;            // 256 x 6144
  float* plin = gh   + (size_t)B_ * 3 * H_;            // 256 x 512
  float* qlin = plin + (size_t)B_ * 512;               // 256 x 512
  float* rf2  = qlin + (size_t)B_ * 512;               // 256 x 256

  // Output slices (tuple order: prior_s, post, h, pmu, pstd, qmu, qstd, rew)
  float* out = (float*)d_out;
  const size_t BT = (size_t)B_ * T_;
  float* o_prior = out;
  float* o_post  = o_prior + BT * L_;
  float* o_h     = o_post  + BT * L_;
  float* o_pmu   = o_h     + BT * H_;
  float* o_pstd  = o_pmu   + BT * L_;
  float* o_qmu   = o_pstd  + BT * L_;
  float* o_qstd  = o_qmu   + BT * L_;
  float* o_rew   = o_qstd  + BT * L_;

  auto conv = [&](const float* src, __bf16* dst, int N, int K, int Npad, int Kpad) {
    long long total = (long long)Npad * Kpad;
    int blocks = (int)((total + 255) / 256);
    convert_pad<<<dim3(blocks), dim3(256), 0, stream>>>(src, dst, N, K, Npad, Kpad);
  };

  // Weight conversion (every call; deterministic)
  conv(W_embed, wsb + oWemb, 2048, 288, 2048, 288);
  conv(W_ih,   wsb + oWih, 6144, 2048, 6144, 2048);
  conv(W_hh,   wsb + oWhh, 6144, 2048, 6144, 2048);
  conv(Wp1,    wsb + oWp1, 200, 2048, NP_, 2048);
  conv(Wp2,    wsb + oWp2, 200, 200, NP_, NP_);
  conv(Wpm,    wsb + oWph, 256, 200, 256, NP_);
  conv(Wps,    wsb + oWph + (size_t)256 * NP_, 256, 200, 256, NP_);
  conv(Wq1,    wsb + oWq1, 200, 3072, NP_, 3072);
  conv(Wq2,    wsb + oWq2, 200, 200, NP_, NP_);
  conv(Wqm,    wsb + oWqh, 256, 200, 256, NP_);
  conv(Wqs,    wsb + oWqh + (size_t)256 * NP_, 256, 200, 256, NP_);
  conv(Wr1,    wsb + oWr1, 200, 2304, NP_, 2304);
  conv(Wr2,    wsb + oWr2, 200, 200, NP_, NP_);

  // Initial carry: post (bf16), h (fp32 + bf16)
  cvt_f32_bf16<<<dim3((B_ * L_ + 255) / 256), dim3(256), 0, stream>>>(prevS, post_bf, B_ * L_);
  copy_f32<<<dim3((B_ * H_ + 255) / 256), dim3(256), 0, stream>>>(prevH, h_f, B_ * H_);
  cvt_f32_bf16<<<dim3((B_ * H_ + 255) / 256), dim3(256), 0, stream>>>(prevH, h_bf, B_ * H_);

  auto gemm = [&](const __bf16* Aop, const __bf16* Wop, int K, int Npad,
                  const float* bias, int Nbias,
                  float* OutF, __bf16* OutB, int sOut, int act) {
    dim3 g(B_ / 128, Npad / 128);
    gemm_bf16<<<g, dim3(256), 0, stream>>>(Aop, Wop, K, bias, Nbias,
                                           OutF, OutB, sOut, act);
  };

  const int eb = (B_ * H_ + 255) / 256;   // elementwise grid for B*H
  const int el = (B_ * L_ + 255) / 256;   // elementwise grid for B*L

  for (int t = 0; t < T_; ++t) {
    // xcat = [post | act_t] (bf16)
    cat_bf_f32<<<dim3((B_ * (L_ + A_) + 255) / 256), dim3(256), 0, stream>>>(
        post_bf, L_, acts + (size_t)t * A_, T_ * A_, A_, xcat, B_);
    // emb = relu(xcat @ W_embed^T + b)                   M=256,K=288,N=2048
    gemm(xcat, wsb + oWemb, 288, 2048, b_embed, 2048, nullptr, emb_bf, 2048, 1);
    // gi = emb @ W_ih^T + b_ih ; gh = h @ W_hh^T + b_hh  K=2048,N=6144
    gemm(emb_bf, wsb + oWih, 2048, 6144, b_ih, 6144, gi, nullptr, 6144, 0);
    gemm(h_bf,   wsb + oWhh, 2048, 6144, b_hh, 6144, gh, nullptr, 6144, 0);
    // GRU gates -> h_f (in place), h_bf, out h[B,T,H]
    gru_gate<<<dim3(eb), dim3(256), 0, stream>>>(gi, gh, h_f, h_bf, o_h, t);
    // prior MLP
    gemm(h_bf,   wsb + oWp1, 2048, NP_, bp1, 200, nullptr, pf1_bf, NP_, 1);
    gemm(pf1_bf, wsb + oWp2, NP_,  NP_, bp2, 200, nullptr, pf_bf,  NP_, 1);
    gemm(pf_bf,  wsb + oWph, NP_,  512, nullptr, 0, plin, nullptr, 512, 0);
    head_elem<<<dim3(el), dim3(256), 0, stream>>>(
        plin, bpm, bps, nzp, t, o_pmu, o_pstd, o_prior, nullptr);
    // posterior MLP: qcat = [h | enc_t], K=3072
    cat_bf_f32<<<dim3((B_ * (H_ + E_) + 255) / 256), dim3(256), 0, stream>>>(
        h_bf, H_, enc + (size_t)t * E_, T_ * E_, E_, qcat, B_);
    gemm(qcat,   wsb + oWq1, 3072, NP_, bq1, 200, nullptr, qf1_bf, NP_, 1);
    gemm(qf1_bf, wsb + oWq2, NP_,  NP_, bq2, 200, nullptr, qf_bf,  NP_, 1);
    gemm(qf_bf,  wsb + oWqh, NP_,  512, nullptr, 0, qlin, nullptr, 512, 0);
    head_elem<<<dim3(el), dim3(256), 0, stream>>>(
        qlin, bqm, bqs, nzq, t, o_qmu, o_qstd, o_post, post_bf);
    // reward: rcat = [post | h], K=2304
    cat_bf_bf<<<dim3((B_ * (L_ + H_) + 255) / 256), dim3(256), 0, stream>>>(
        post_bf, L_, h_bf, H_, rcat, B_);
    gemm(rcat,   wsb + oWr1, 2304, NP_, br1, 200, nullptr, rf1_bf, NP_, 1);
    gemm(rf1_bf, wsb + oWr2, NP_,  NP_, br2, 200, rf2, nullptr, NP_, 1);
    reward_dot<<<dim3(1), dim3(256), 0, stream>>>(rf2, Wr3, br3, o_rew, t);
  }
}
